// AR_GAS_Net_89189290869336
// MI455X (gfx1250) — matrix operand order repaired and checked by reading, not compiled, not run on hardware
//
#include <hip/hip_runtime.h>
#include <math.h>

// ---------------- problem constants ----------------
#define D_IN  200
#define HDIM  100
#define KOUT  64
#define XKP   232   // padded K stride (f16) for x tile / W1^T (224 used + 8 pad -> bank-friendly)
#define HKP   136   // padded K stride (f16) for h tiles / W2^T / W3^T (128 used + 8 pad)
#define DPW   68    // deep_preds stride in f32 (fits inside per-wave scratch)
#define ROWS_W 16   // batch rows per wave (one WMMA M tile)
#define WAVES  8
#define BROWS (ROWS_W * WAVES)  // 128 rows per block

typedef _Float16 half8  __attribute__((ext_vector_type(8)));
typedef _Float16 half16 __attribute__((ext_vector_type(16)));
typedef float    f32x8  __attribute__((ext_vector_type(8)));

// A-matrix 16x32 f16 fragment (ISA 7.12.2):
// lanes 0-15: M=lane, slots0-7 -> K=k0..k0+7,   slots8-15 -> K=k0+16..k0+23
// lanes16-31: M=lane-16, slots0-7 -> K=k0+8..15, slots8-15 -> K=k0+24..31
__device__ inline half16 frag_a(const _Float16* base, int stride, int m, int hf, int k0) {
    const _Float16* row = base + m * stride + k0 + hf * 8;
    half8 lo = *(const half8*)(row);
    half8 hi = *(const half8*)(row + 16);
    return __builtin_shufflevector(lo, hi, 0,1,2,3,4,5,6,7,8,9,10,11,12,13,14,15);
}

// B-matrix 32x16 f16 fragment from W^T (K contiguous per column):
// lanes 0-15: N=lane,   slots0-7 -> K=k0..k0+7,   slots8-15 -> K=k0+16..k0+23
// lanes16-31: N=lane-16, slots0-7 -> K=k0+8..15, slots8-15 -> K=k0+24..31
__device__ inline half16 frag_b(const _Float16* wt, int stride, int n, int hf, int k0) {
    const _Float16* p = wt + n * stride + k0 + hf * 8;
    half8 lo = *(const half8*)(p);
    half8 hi = *(const half8*)(p + 16);
    return __builtin_shufflevector(lo, hi, 0,1,2,3,4,5,6,7,8,9,10,11,12,13,14,15);
}

__global__ __launch_bounds__(256)
__attribute__((amdgpu_waves_per_eu(4)))
void ar_gas_fused(const float* __restrict__ x,
                  const float* __restrict__ last_mu,
                  const float* __restrict__ last_sigma,
                  const float* p_amu, const float* p_as,
                  const float* p_bmu, const float* p_bs,
                  const float* p_omu, const float* p_os,
                  const float* p_nu,  const float* p_ns,
                  const float* __restrict__ W1, const float* __restrict__ b1,
                  const float* __restrict__ W2, const float* __restrict__ b2,
                  const float* __restrict__ W3, const float* __restrict__ b3,
                  float* __restrict__ out, int Bn)
{
    __shared__ __align__(16) _Float16 sW1T[112 * XKP];          // 51968 B
    __shared__ __align__(16) _Float16 sW2T[112 * HKP];          // 30464 B
    __shared__ __align__(16) _Float16 sW3T[ 64 * HKP];          // 17408 B
    __shared__ __align__(16) _Float16 sScr[WAVES][ROWS_W * XKP];// 59392 B  (total ~155.5 KB)

    const int tid  = threadIdx.x;
    const int wv   = tid >> 5;
    const int lane = tid & 31;
    const int hf   = lane >> 4;
    const int m    = lane & 15;

    // uniform GAS scalars
    const float a_mu = p_amu[0], a_s = p_as[0];
    const float b_mu = p_bmu[0], b_s = p_bs[0];
    const float o_mu = p_omu[0], o_s = p_os[0];
    const float nu   = p_nu[0],  ns  = p_ns[0];
    const float wnu  = 1.0f + 1.0f / nu;

    // ---- stage transposed, zero-padded f16 weights into LDS (block-wide) ----
    for (int i = tid; i < 112 * XKP; i += 256) {
        int n = i / XKP, k = i - n * XKP;
        float v = (n < HDIM && k < D_IN) ? W1[k * HDIM + n] : 0.0f;
        sW1T[i] = (_Float16)v;
    }
    for (int i = tid; i < 112 * HKP; i += 256) {
        int n = i / HKP, k = i - n * HKP;
        float v = (n < HDIM && k < HDIM) ? W2[k * HDIM + n] : 0.0f;
        sW2T[i] = (_Float16)v;
    }
    for (int i = tid; i < 64 * HKP; i += 256) {
        int n = i / HKP, k = i - n * HKP;
        float v = (k < HDIM) ? W3[k * KOUT + n] : 0.0f;
        sW3T[i] = (_Float16)v;
    }

    // ---- stage this wave's 16 x rows into its scratch region (f32 -> f16, zero pad K) ----
    _Float16* scr = sScr[wv];
    const long rowbase = (long)blockIdx.x * BROWS + wv * ROWS_W;
    for (int i = lane; i < ROWS_W * XKP; i += 32) {
        int r = i / XKP, k = i - r * XKP;
        long row = rowbase + r;
        float v = (k < D_IN && row < Bn) ? x[row * (long)D_IN + k] : 0.0f;
        scr[i] = (_Float16)v;
    }
    __syncthreads();

    // ================= GEMM1: [16 x 224] * [224 x 112] =================
    f32x8 acc1[7] = {};
    #pragma unroll
    for (int ks = 0; ks < 7; ++ks) {
        half16 a = frag_a(scr, XKP, m, hf, ks * 32);
        #pragma unroll
        for (int nt = 0; nt < 7; ++nt) {
            half16 b = frag_b(sW1T, XKP, nt * 16 + m, hf, ks * 32);
            acc1[nt] = __builtin_amdgcn_wmma_f32_16x16x32_f16(
                false, a, false, b, (short)0, acc1[nt], false, false);
        }
    }

    // bias + relu, store h1 (f16) into scratch (reuse; wave-local DS ordering keeps this safe)
    _Float16* hbuf = scr;
    #pragma unroll
    for (int nt = 0; nt < 7; ++nt) {
        int n = nt * 16 + m;
        float bv = (n < HDIM) ? b1[n] : 0.0f;
        #pragma unroll
        for (int r = 0; r < 8; ++r) {
            float v = acc1[nt][r] + bv;
            v = v > 0.0f ? v : 0.0f;
            hbuf[(r + hf * 8) * HKP + n] = (_Float16)v;
        }
    }
    for (int i = lane; i < ROWS_W * 24; i += 32) {   // zero K = 112..135
        int r = i / 24, k = 112 + (i - r * 24);
        hbuf[r * HKP + k] = (_Float16)0.0f;
    }

    // ================= GEMM2: [16 x 128] * [128 x 112] =================
    f32x8 acc2[7] = {};
    #pragma unroll
    for (int ks = 0; ks < 4; ++ks) {
        half16 a = frag_a(hbuf, HKP, m, hf, ks * 32);
        #pragma unroll
        for (int nt = 0; nt < 7; ++nt) {
            half16 b = frag_b(sW2T, HKP, nt * 16 + m, hf, ks * 32);
            acc2[nt] = __builtin_amdgcn_wmma_f32_16x16x32_f16(
                false, a, false, b, (short)0, acc2[nt], false, false);
        }
    }
    #pragma unroll
    for (int nt = 0; nt < 7; ++nt) {
        int n = nt * 16 + m;
        float bv = (n < HDIM) ? b2[n] : 0.0f;
        #pragma unroll
        for (int r = 0; r < 8; ++r) {
            float v = acc2[nt][r] + bv;
            v = v > 0.0f ? v : 0.0f;
            hbuf[(r + hf * 8) * HKP + n] = (_Float16)v;
        }
    }
    for (int i = lane; i < ROWS_W * 24; i += 32) {
        int r = i / 24, k = 112 + (i - r * 24);
        hbuf[r * HKP + k] = (_Float16)0.0f;
    }

    // ================= GEMM3: [16 x 128] * [128 x 64] =================
    f32x8 acc3[4] = {};
    #pragma unroll
    for (int ks = 0; ks < 4; ++ks) {
        half16 a = frag_a(hbuf, HKP, m, hf, ks * 32);
        #pragma unroll
        for (int nt = 0; nt < 4; ++nt) {
            half16 b = frag_b(sW3T, HKP, nt * 16 + m, hf, ks * 32);
            acc3[nt] = __builtin_amdgcn_wmma_f32_16x16x32_f16(
                false, a, false, b, (short)0, acc3[nt], false, false);
        }
    }

    // deep_preds (+b3) as f32 into scratch
    float* dpb = (float*)scr;
    #pragma unroll
    for (int nt = 0; nt < 4; ++nt) {
        int n = nt * 16 + m;
        float bv = b3[n];
        #pragma unroll
        for (int r = 0; r < 8; ++r)
            dpb[(r + hf * 8) * DPW + n] = acc3[nt][r] + bv;
    }

    // ---- Student-t GAS recurrence: one row per lane (lanes 0-15) ----
    if (lane < 16) {
        long row = rowbase + lane;
        if (row < Bn) {
            float mu = last_mu[row];
            float s2 = last_sigma[row];
            float* dpr = dpb + lane * DPW;
            for (int k = 0; k < KOUT; ++k) {
                float y = dpr[k];
                float e = y - mu;
                float denom = 1.0f + e * e / (nu * s2);
                mu = o_mu + ns * a_mu * (wnu * e / denom) + b_mu * mu;
                s2 = o_s  + ns * a_s  * (wnu * e * e / denom - s2) + b_s * s2;
                dpr[k] = y * sqrtf(s2) + mu;       // deep_preds*sigma + mu
            }
        }
    }

    // ---- coalesced float4 store of this wave's 16x64 output block ----
    #pragma unroll
    for (int i = 0; i < 8; ++i) {
        int idx = lane * 4 + i * 128;
        int r = idx >> 6, c = idx & 63;
        long row = rowbase + r;
        if (row < Bn) {
            float4 v = *(const float4*)(dpb + r * DPW + c);
            *(float4*)(out + row * (long)KOUT + c) = v;
        }
    }
}

extern "C" void kernel_launch(void* const* d_in, const int* in_sizes, int n_in,
                              void* d_out, int out_size, void* d_ws, size_t ws_size,
                              hipStream_t stream) {
    const float* x    = (const float*)d_in[0];
    const float* lmu  = (const float*)d_in[1];
    const float* lsig = (const float*)d_in[2];
    const float* amu  = (const float*)d_in[3];
    const float* asg  = (const float*)d_in[4];
    const float* bmu  = (const float*)d_in[5];
    const float* bsg  = (const float*)d_in[6];
    const float* omu  = (const float*)d_in[7];
    const float* osg  = (const float*)d_in[8];
    const float* nu   = (const float*)d_in[9];
    const float* nstr = (const float*)d_in[10];
    const float* W1   = (const float*)d_in[11];
    const float* b1   = (const float*)d_in[12];
    const float* W2   = (const float*)d_in[13];
    const float* b2   = (const float*)d_in[14];
    const float* W3   = (const float*)d_in[15];
    const float* b3   = (const float*)d_in[16];

    const int Bn   = in_sizes[0] / D_IN;
    const int grid = (Bn + BROWS - 1) / BROWS;

    ar_gas_fused<<<grid, 256, 0, stream>>>(x, lmu, lsig,
                                           amu, asg, bmu, bsg, omu, osg, nu, nstr,
                                           W1, b1, W2, b2, W3, b3,
                                           (float*)d_out, Bn);
}
